// BiShowAttenTellModel_57990648430748
// MI455X (gfx1250) — compile-verified, matrix operand order checked by reading, not run on hardware
//
#include <hip/hip_runtime.h>
#include <cstdint>
#include <cstddef>

// ---- model dims (from reference) ----
#define BB    128     // batch
#define LL    196     // attention locations
#define EE    512     // embed size
#define RR    512     // rnn size
#define AHS   512     // attention hidden
#define FCC   2048    // fc feat size
#define AFF   2048    // att feat size
#define SS    16      // seq len
#define VV1   9488    // vocab + 1
#define NSTEP 18      // scan steps (fc_e, bos, 16 tokens)
#define NTOP  17      // output steps
#define KCAT  1536    // [x | h | att_res]
#define NGATE 2048    // 4 * RR

typedef __bf16 bf16;
typedef __attribute__((ext_vector_type(16))) __bf16 v16bf;
typedef __attribute__((ext_vector_type(8)))  float  v8f;

// pack two f32 -> two bf16 (RNE) in one u32 (low = x, high = y)
__device__ __forceinline__ unsigned pack2bf(float x, float y) {
  unsigned ux = __builtin_bit_cast(unsigned, x);
  unsigned uy = __builtin_bit_cast(unsigned, y);
  ux += 0x7FFFu + ((ux >> 16) & 1u);
  uy += 0x7FFFu + ((uy >> 16) & 1u);
  return (ux >> 16) | (uy & 0xFFFF0000u);
}
__device__ __forceinline__ float sigmoidf(float x) { return 1.f / (1.f + __expf(-x)); }

// =====================================================================
// Generic GEMM: C(MxN) = A(MxK) * B(KxN) [+ bias] [prelu]
// f32 global operands -> bf16 LDS tiles -> v_wmma_f32_16x16x32_bf16.
// Block tile 128x128x32, 8 waves (wave32), each wave 32(M)x64(N).
// Both LDS tiles are kept "fragment-major" (A: M-major, B: N-major via a
// register 4x4 transpose at staging) so each WMMA fragment is exactly two
// contiguous 16B runs per lane -> 2 x ds_load_b128, no sub-word shuffling.
// =====================================================================
#define BM 128
#define BN 128
#define BK 32
#define TSTR 20     // u32 row stride for LDS tiles (16 data + 4 pad)
enum { EPI_BIAS = 1, EPI_PRELU = 2 };

struct q2 { uint4 a, b; };

// frag word i holds bf16 pair K=2i,2i+1 of the ISA 16-bit 16x32 layout:
// words [half*4 .. half*4+3] then [8+half*4 .. +3]  -> two uint4 loads
__device__ __forceinline__ v16bf load_frag(const unsigned* __restrict__ row, int half) {
  q2 t;
  t.a = *(const uint4*)(row + half * 4);
  t.b = *(const uint4*)(row + 8 + half * 4);
  return __builtin_bit_cast(v16bf, t);
}

__global__ __launch_bounds__(256)
void gemm_f32_wmma_bf16(const float* __restrict__ A, const float* __restrict__ Bm,
                        const float* __restrict__ bias, const float* __restrict__ prelu_a,
                        float* __restrict__ C, int M, int N, int K, int flags)
{
  __shared__ unsigned sA [BM * TSTR];   // A tile, M-major, bf16 pairs in u32
  __shared__ unsigned sBT[BN * TSTR];   // B tile, N-major (transposed), bf16 pairs

  const int tid  = threadIdx.x;
  const int wid  = tid >> 5;
  const int lane = tid & 31;
  const int half = lane >> 4;
  const int l16  = lane & 15;

  const int bm = blockIdx.y * BM;
  const int bn = blockIdx.x * BN;
  const int wm = (wid >> 1) * 32;   // 4 wave-rows
  const int wn = (wid & 1) * 64;    // 2 wave-cols

  v8f zero = {};
  v8f acc[2][4];
#pragma unroll
  for (int i = 0; i < 2; ++i)
#pragma unroll
    for (int j = 0; j < 4; ++j) acc[i][j] = zero;

  const int ktiles = (K + BK - 1) / BK;
  for (int kt = 0; kt < ktiles; ++kt) {
    const int k0 = kt * BK;

    // ---- stage A tile (128x32 f32 -> bf16 pairs), 4 float4 per thread ----
#pragma unroll
    for (int i = 0; i < 4; ++i) {
      int f = tid + 256 * i;            // 0..1023 float4 slots
      int r = f >> 3;                   // row (8 float4 per row)
      int c = (f & 7) << 2;             // col 0,4,..,28
      float4 v = {0.f, 0.f, 0.f, 0.f};
      int gm = bm + r;
      if (gm < M && (k0 + c + 3) < K)
        v = *(const float4*)(A + (size_t)gm * K + k0 + c);
      *(uint2*)(sA + (size_t)r * TSTR + (c >> 1)) =
          make_uint2(pack2bf(v.x, v.y), pack2bf(v.z, v.w));
    }
    // ---- stage B tile transposed (32x128 -> N-major), 4x4 block/thread ----
    {
      int bcol = tid & 31;              // 32 column groups of 4
      int brow = tid >> 5;              // 8 row groups of 4
      int gn = bn + bcol * 4;
      int kk = brow * 4;
      const float* s = Bm + (size_t)(k0 + kk) * N + gn;
      float rr[4][4];
      if (gn + 3 < N) {
#pragma unroll
        for (int j = 0; j < 4; ++j) {
          float4 v = *(const float4*)(s + (size_t)j * N);
          rr[j][0] = v.x; rr[j][1] = v.y; rr[j][2] = v.z; rr[j][3] = v.w;
        }
      } else {
#pragma unroll
        for (int j = 0; j < 4; ++j)
#pragma unroll
          for (int jj = 0; jj < 4; ++jj)
            rr[j][jj] = (gn + jj < N) ? s[(size_t)j * N + jj] : 0.f;
      }
#pragma unroll
      for (int j = 0; j < 4; ++j) {     // column n+j, K run kk..kk+3
        *(uint2*)(sBT + (size_t)(bcol * 4 + j) * TSTR + (kk >> 1)) =
            make_uint2(pack2bf(rr[0][j], rr[1][j]), pack2bf(rr[2][j], rr[3][j]));
      }
    }
    __syncthreads();

    if (kt + 1 < ktiles) {  // CDNA5 global_prefetch_b8 hints for next K-tile
      __builtin_prefetch(A + (size_t)(bm + (tid >> 1)) * K + k0 + BK, 0, 1);
      __builtin_prefetch(Bm + (size_t)(k0 + BK + (tid >> 3)) * N + bn, 0, 1);
    }

    // ---- fragment loads: 2 x ds_load_b128 each ----
    v16bf af[2], bfv[4];
#pragma unroll
    for (int tm = 0; tm < 2; ++tm)
      af[tm] = load_frag(sA + (size_t)(wm + tm * 16 + l16) * TSTR, half);
#pragma unroll
    for (int tn = 0; tn < 4; ++tn)
      bfv[tn] = load_frag(sBT + (size_t)(wn + tn * 16 + l16) * TSTR, half);

#pragma unroll
    for (int tm = 0; tm < 2; ++tm)
#pragma unroll
      for (int tn = 0; tn < 4; ++tn)
        acc[tm][tn] = __builtin_amdgcn_wmma_f32_16x16x32_bf16(
            false, af[tm], false, bfv[tn], (short)0, acc[tm][tn], false, false);
    __syncthreads();
  }

  // ---- epilogue: C/D layout (VGPR r -> m = half*8 + r, n = l16) ----
  const float apr = (flags & EPI_PRELU) ? prelu_a[0] : 0.f;
#pragma unroll
  for (int tm = 0; tm < 2; ++tm)
#pragma unroll
    for (int tn = 0; tn < 4; ++tn) {
      int n = bn + wn + tn * 16 + l16;
      if (n >= N) continue;
      float bv = (flags & EPI_BIAS) ? bias[n] : 0.f;
      int mb = bm + wm + tm * 16 + half * 8;
#pragma unroll
      for (int r = 0; r < 8; ++r) {
        int m = mb + r;
        if (m >= M) continue;
        float v = acc[tm][tn][r] + bv;
        if (flags & EPI_PRELU) v = (v > 0.f) ? v : apr * v;
        C[(size_t)m * N + n] = v;
      }
    }
}

// =====================================================================
// Attention: e = tanh(att_pre + h_att) . w_alpha + b_alpha ; softmax(L);
// att_res = sum_l alpha_l * att_e[:,l,:]. One block per batch element.
// =====================================================================
__global__ __launch_bounds__(256)
void attention_kernel(const float* __restrict__ att_pre, const float* __restrict__ h_att,
                      const float* __restrict__ att_e, const float* __restrict__ w_alpha,
                      const float* __restrict__ b_alpha, float* __restrict__ att_res)
{
  __shared__ float e_s[LL + 4];
  __shared__ float red[8];
  const int b = blockIdx.x;
  const int tid = threadIdx.x, wid = tid >> 5, lane = tid & 31;
  const float* hb = h_att + (size_t)b * AHS;

  // phase 1: one wave per l, lane-strided dot over AH=512
  for (int l = wid; l < LL; l += 8) {
    const float* ap = att_pre + ((size_t)b * LL + l) * AHS;
    float acc = 0.f;
    for (int k = lane; k < AHS; k += 32)
      acc += tanhf(ap[k] + hb[k]) * w_alpha[k];
#pragma unroll
    for (int off = 16; off > 0; off >>= 1) acc += __shfl_xor(acc, off, 32);
    if (lane == 0) e_s[l] = acc + b_alpha[0];
  }
  __syncthreads();

  // phase 2: softmax over L=196
  float v = (tid < LL) ? e_s[tid] : -1e30f;
#pragma unroll
  for (int off = 16; off > 0; off >>= 1) v = fmaxf(v, __shfl_xor(v, off, 32));
  if (lane == 0) red[wid] = v;
  __syncthreads();
  float mx = red[0];
#pragma unroll
  for (int i = 1; i < 8; ++i) mx = fmaxf(mx, red[i]);
  __syncthreads();
  float s = 0.f;
  if (tid < LL) { float ex = __expf(e_s[tid] - mx); e_s[tid] = ex; s = ex; }
#pragma unroll
  for (int off = 16; off > 0; off >>= 1) s += __shfl_xor(s, off, 32);
  if (lane == 0) red[wid] = s;
  __syncthreads();
  float ssum = 0.f;
#pragma unroll
  for (int i = 0; i < 8; ++i) ssum += red[i];
  const float inv = 1.f / ssum;

  // phase 3: att_res[b, col] = sum_l alpha * att_e
  for (int col = tid; col < RR; col += 256) {
    const float* ae = att_e + (size_t)b * LL * RR + col;
    float acc = 0.f;
    for (int l = 0; l < LL; ++l) acc += e_s[l] * ae[(size_t)l * RR];
    att_res[(size_t)b * RR + col] = acc * inv;
  }
}

// ---- pack [x | h | att_res] into A_cat (B x 1536); x from fc_e / emb ----
__global__ __launch_bounds__(256)
void pack_xha_kernel(const float* __restrict__ embW, const int* __restrict__ labels,
                     const float* __restrict__ fc_e, const float* __restrict__ h,
                     const float* __restrict__ att_res, int t, int backward,
                     float* __restrict__ Acat)
{
  int idx = blockIdx.x * 256 + threadIdx.x;
  if (idx >= BB * KCAT) return;
  int b = idx / KCAT, col = idx % KCAT;
  float v;
  if (col < EE) {
    if (t == 0)       v = fc_e[(size_t)b * EE + col];
    else if (t == 1)  v = embW[col];                       // BOS row 0
    else {
      int j = t - 2;
      if (backward) j = (30 - j) & 15;                     // idxs = (16-(j+2)) mod 16
      int lab = labels[b * SS + j];
      v = embW[(size_t)lab * EE + col];
    }
  } else if (col < 2 * EE) v = h[(size_t)b * RR + (col - EE)];
  else                     v = att_res[(size_t)b * RR + (col - 2 * EE)];
  Acat[idx] = v;
}

// ---- LSTM pointwise update; i,f,o,g split of gates ----
__global__ __launch_bounds__(256)
void lstm_step_kernel(const float* __restrict__ gates, float* __restrict__ cstate,
                      float* __restrict__ hstate, float* __restrict__ hs_out)
{
  int idx = blockIdx.x * 256 + threadIdx.x;
  if (idx >= BB * RR) return;
  int b = idx >> 9, n = idx & (RR - 1);
  const float* g = gates + (size_t)b * NGATE;
  float ig = sigmoidf(g[n]);
  float fg = sigmoidf(g[RR + n]);
  float og = sigmoidf(g[2 * RR + n]);
  float gg = tanhf(g[3 * RR + n]);
  float c2 = fg * cstate[idx] + ig * gg;
  float h2 = og * tanhf(c2);
  cstate[idx] = c2;
  hstate[idx] = h2;
  hs_out[idx] = h2;
}

// ---- comb[t] = hs_f[t+1] + hs_b[17-t] ----
__global__ __launch_bounds__(256)
void comb_kernel(const float* __restrict__ hsf, const float* __restrict__ hsb,
                 float* __restrict__ comb)
{
  int idx = blockIdx.x * 256 + threadIdx.x;
  if (idx >= NTOP * BB * RR) return;
  int t = idx / (BB * RR), rem = idx % (BB * RR);
  comb[idx] = hsf[(size_t)(t + 1) * BB * RR + rem] + hsb[(size_t)(NTOP - t) * BB * RR + rem];
}

// ---- pack [Wi; Uh; Wz] -> Wcat (1536 x 2048) ----
__global__ __launch_bounds__(256)
void pack_w_kernel(const float* __restrict__ Wi, const float* __restrict__ Uh,
                   const float* __restrict__ Wz, float* __restrict__ Wcat)
{
  int i = blockIdx.x * 256 + threadIdx.x;
  if (i >= RR * NGATE) return;
  Wcat[i]                           = Wi[i];
  Wcat[(size_t)RR * NGATE + i]      = Uh[i];
  Wcat[(size_t)2 * RR * NGATE + i]  = Wz[i];
}

__global__ __launch_bounds__(256)
void copy4_kernel(const float* __restrict__ src, float* d0, float* d1, float* d2, float* d3, int n)
{
  int i = blockIdx.x * 256 + threadIdx.x;
  if (i < n) { float v = src[i]; d0[i] = v; d1[i] = v; d2[i] = v; d3[i] = v; }
}

// ---- row log-softmax over vocab, transposed store to (B, 17, V1) ----
__global__ __launch_bounds__(256)
void log_softmax_kernel(const float* __restrict__ logits, float* __restrict__ out)
{
  __shared__ float red[8];
  const int row = blockIdx.x;                // t*128 + b
  const int t = row >> 7, b = row & 127;
  const float* x = logits + (size_t)row * VV1;
  float* o = out + ((size_t)b * NTOP + t) * VV1;
  const int tid = threadIdx.x, lane = tid & 31, wid = tid >> 5;

  float mx = -1e30f;
  for (int i = tid; i < VV1; i += 256) mx = fmaxf(mx, x[i]);
#pragma unroll
  for (int off = 16; off > 0; off >>= 1) mx = fmaxf(mx, __shfl_xor(mx, off, 32));
  if (lane == 0) red[wid] = mx;
  __syncthreads();
  float m = red[0];
#pragma unroll
  for (int i = 1; i < 8; ++i) m = fmaxf(m, red[i]);
  __syncthreads();
  float s = 0.f;
  for (int i = tid; i < VV1; i += 256) s += __expf(x[i] - m);
#pragma unroll
  for (int off = 16; off > 0; off >>= 1) s += __shfl_xor(s, off, 32);
  if (lane == 0) red[wid] = s;
  __syncthreads();
  float sum = 0.f;
#pragma unroll
  for (int i = 0; i < 8; ++i) sum += red[i];
  const float lse = m + __logf(sum);
  for (int i = tid; i < VV1; i += 256) o[i] = x[i] - lse;
}

// =====================================================================
extern "C" void kernel_launch(void* const* d_in, const int* in_sizes, int n_in,
                              void* d_out, int out_size, void* d_ws, size_t ws_size,
                              hipStream_t stream)
{
  (void)in_sizes; (void)n_in; (void)out_size; (void)ws_size;
  const float* fc_feats  = (const float*)d_in[0];
  const float* att_feats = (const float*)d_in[1];
  const int*   labels    = (const int*)d_in[2];
  const float* embW      = (const float*)d_in[3];
  const float* img_W     = (const float*)d_in[4];
  const float* img_b     = (const float*)d_in[5];
  const float* att_W     = (const float*)d_in[6];
  const float* att_b     = (const float*)d_in[7];
  const float* proj_W    = (const float*)d_in[8];
  const float* proj_b    = (const float*)d_in[9];
  const float* prelu_a   = (const float*)d_in[10];
  const float* Wa[2]  = {(const float*)d_in[11], (const float*)d_in[21]};
  const float* ba[2]  = {(const float*)d_in[12], (const float*)d_in[22]};
  const float* Wh[2]  = {(const float*)d_in[13], (const float*)d_in[23]};
  const float* bh[2]  = {(const float*)d_in[14], (const float*)d_in[24]};
  const float* wal[2] = {(const float*)d_in[15], (const float*)d_in[25]};
  const float* bal[2] = {(const float*)d_in[16], (const float*)d_in[26]};
  const float* Wi_[2] = {(const float*)d_in[17], (const float*)d_in[27]};
  const float* Uh_[2] = {(const float*)d_in[18], (const float*)d_in[28]};
  const float* Wz_[2] = {(const float*)d_in[19], (const float*)d_in[29]};
  const float* bg[2]  = {(const float*)d_in[20], (const float*)d_in[30]};

  size_t off = 0;
  auto alloc = [&](size_t bytes) -> float* {
    off = (off + 255) & ~(size_t)255;
    float* p = (float*)((char*)d_ws + off);
    off += bytes;
    return p;
  };
  float* fc_e       = alloc((size_t)BB * RR * 4);
  float* att_e      = alloc((size_t)BB * LL * RR * 4);
  float* att_pre[2] = {alloc((size_t)BB * LL * AHS * 4), alloc((size_t)BB * LL * AHS * 4)};
  float* Wcat[2]    = {alloc((size_t)KCAT * NGATE * 4), alloc((size_t)KCAT * NGATE * 4)};
  float* hst[2]     = {alloc((size_t)BB * RR * 4), alloc((size_t)BB * RR * 4)};
  float* cst[2]     = {alloc((size_t)BB * RR * 4), alloc((size_t)BB * RR * 4)};
  float* hs[2]      = {alloc((size_t)NSTEP * BB * RR * 4), alloc((size_t)NSTEP * BB * RR * 4)};
  float* h_att      = alloc((size_t)BB * AHS * 4);
  float* att_res    = alloc((size_t)BB * RR * 4);
  float* Acat       = alloc((size_t)BB * KCAT * 4);
  float* gates      = alloc((size_t)BB * NGATE * 4);
  float* comb       = alloc((size_t)NTOP * BB * RR * 4);
  float* logits     = alloc((size_t)NTOP * BB * VV1 * 4);

  dim3 blk(256);

  // 1) fc_e = prelu(fc_feats @ img_W + img_b)
  gemm_f32_wmma_bf16<<<dim3(RR / 128, BB / 128), blk, 0, stream>>>(
      fc_feats, img_W, img_b, prelu_a, fc_e, BB, RR, FCC, EPI_BIAS | EPI_PRELU);
  // 2) att_e = prelu(att_feats @ att_W + att_b), M = B*L = 25088
  gemm_f32_wmma_bf16<<<dim3(RR / 128, (BB * LL) / 128), blk, 0, stream>>>(
      att_feats, att_W, att_b, prelu_a, att_e, BB * LL, RR, AFF, EPI_BIAS | EPI_PRELU);
  // 3) att_pre[c] = att_e @ Wa_att + ba_att
  for (int c = 0; c < 2; ++c)
    gemm_f32_wmma_bf16<<<dim3(AHS / 128, (BB * LL) / 128), blk, 0, stream>>>(
        att_e, Wa[c], ba[c], prelu_a, att_pre[c], BB * LL, AHS, EE, EPI_BIAS);
  // 4) Wcat[c] = [Wi; Uh; Wz]
  for (int c = 0; c < 2; ++c)
    pack_w_kernel<<<(RR * NGATE + 255) / 256, blk, 0, stream>>>(Wi_[c], Uh_[c], Wz_[c], Wcat[c]);
  // 5) h0 = c0 = fc_e for both cores
  copy4_kernel<<<(BB * RR + 255) / 256, blk, 0, stream>>>(fc_e, hst[0], cst[0], hst[1], cst[1], BB * RR);

  // 6) recurrence: 18 steps per core (core1 = backward token order)
  for (int c = 0; c < 2; ++c) {
    for (int t = 0; t < NSTEP; ++t) {
      gemm_f32_wmma_bf16<<<dim3(AHS / 128, 1), blk, 0, stream>>>(
          hst[c], Wh[c], bh[c], prelu_a, h_att, BB, AHS, RR, EPI_BIAS);
      attention_kernel<<<BB, blk, 0, stream>>>(att_pre[c], h_att, att_e, wal[c], bal[c], att_res);
      pack_xha_kernel<<<(BB * KCAT + 255) / 256, blk, 0, stream>>>(
          embW, labels, fc_e, hst[c], att_res, t, c, Acat);
      gemm_f32_wmma_bf16<<<dim3(NGATE / 128, 1), blk, 0, stream>>>(
          Acat, Wcat[c], bg[c], prelu_a, gates, BB, NGATE, KCAT, EPI_BIAS);
      lstm_step_kernel<<<(BB * RR + 255) / 256, blk, 0, stream>>>(
          gates, cst[c], hst[c], hs[c] + (size_t)t * BB * RR);
    }
  }

  // 7) comb[t] = hs_f[t+1] + hs_b[17-t]
  comb_kernel<<<(NTOP * BB * RR + 255) / 256, blk, 0, stream>>>(hs[0], hs[1], comb);
  // 8) logits = comb @ proj_W + proj_b   (N = 9488, edge-guarded)
  gemm_f32_wmma_bf16<<<dim3((VV1 + 127) / 128, (NTOP * BB) / 128), blk, 0, stream>>>(
      comb, proj_W, proj_b, prelu_a, logits, NTOP * BB, VV1, RR, EPI_BIAS);
  // 9) log_softmax + transpose to (B, 17, V1)
  log_softmax_kernel<<<NTOP * BB, blk, 0, stream>>>(logits, (float*)d_out);
}
